// AtomMoE_25366076850632
// MI455X (gfx1250) — matrix operand order repaired and checked by reading, not compiled, not run on hardware
//
#include <hip/hip_runtime.h>
#include <hip/hip_bf16.h>
#include <stdint.h>

// ---- problem constants (from reference) ----
#define T_TOK   32768      // B*N
#define DFF     512
#define GATEH   256
#define KEXP    8
#define HIDD    2048
#define CAPTY   5120       // ceil(1.25 * T / K)
#define LN_EPS  1e-5f

typedef __attribute__((ext_vector_type(16))) __bf16 v16bf;
typedef __attribute__((ext_vector_type(8)))  float  v8f;

union FragBF { v16bf v; uint4 q[2]; };

__device__ __forceinline__ float gelu_exact(float x) {
    return 0.5f * x * (1.0f + erff(x * 0.70710678118654752f));
}

// ---- CDNA5 async global->LDS path (ASYNCcnt), guarded so we always compile ----
#if defined(__HIP_DEVICE_COMPILE__) && __has_builtin(__builtin_amdgcn_global_load_async_to_lds_b128)
#define ASYNC_LDS 1
#else
#define ASYNC_LDS 0
#endif

#if ASYNC_LDS
// exact parameter type per compiler diagnostic: vector_size(16) int, AS1 / AS3
typedef __attribute__((__vector_size__(4 * sizeof(int)))) int vs4i;
typedef __attribute__((address_space(1))) vs4i* gvs4i_p;
typedef __attribute__((address_space(3))) vs4i* lvs4i_p;

__device__ __forceinline__ void async_cp16(const __bf16* g, __bf16* l) {
    __builtin_amdgcn_global_load_async_to_lds_b128(
        (gvs4i_p)g, (lvs4i_p)l, 0, 0);
}
__device__ __forceinline__ void async_wait0() {
#if __has_builtin(__builtin_amdgcn_s_wait_asynccnt)
    __builtin_amdgcn_s_wait_asynccnt(0);
#else
    asm volatile("s_wait_asynccnt 0x0" ::: "memory");
#endif
}
#endif

// ============================================================================
// Tiled bf16 WMMA GEMM:  C[M,Nn] = A[M,Kk] (bf16) @ BT[Nn,Kk]^T (bf16) + bias
//   EPI 0: store f32 to Cf
//   EPI 1: gelu -> bf16 to Cb
//   EPI 2: scatter f32 rows through slot_tok (capacity-masked) into Cf
// Block tile 64x128, BK=32, 128 threads (4 waves); wave computes 32x64
// (2x4 of 16x16) -> 8 v_wmma per K-step. Async double-buffered LDS staging
// when the toolchain exposes GLOBAL_LOAD_ASYNC_TO_LDS.
// ============================================================================
template<int EPI>
__global__ __launch_bounds__(128)
void gemm_bf16(const __bf16* __restrict__ A, const __bf16* __restrict__ BT,
               const float* __restrict__ bias,
               float* __restrict__ Cf, __bf16* __restrict__ Cb,
               const int* __restrict__ slot_tok, const int* __restrict__ cnt_ptr,
               int M, int Nn, int Kk)
{
    const int tid  = threadIdx.x;
    const int lane = tid & 31;
    const int wave = tid >> 5;
    const int wm   = (wave & 1) * 32;    // wave grid 2x2 over 64x128
    const int wn   = (wave >> 1) * 64;
    const int m0   = blockIdx.x * 64;
    const int n0   = blockIdx.y * 128;
    const int rA   = tid >> 1;           // A: 64 rows, 2 threads/row
    const int sgA  = tid & 1;            // 16-element half of 32-wide K slice

    const v8f vzero = {0.f,0.f,0.f,0.f,0.f,0.f,0.f,0.f};
    v8f acc[2][4];
    #pragma unroll
    for (int i = 0; i < 2; ++i)
        #pragma unroll
        for (int j = 0; j < 4; ++j) acc[i][j] = vzero;

    const int nt = Kk >> 5;

#if ASYNC_LDS
    __shared__ __bf16 sA[2][64 * 40];    // 32 valid + 8 pad per row
    __shared__ __bf16 sB[2][128 * 40];

    // stage one 64x32 A tile + 128x32 B tile into LDS buffer `buf`
    auto issue = [&](int buf, int kt) {
        const __bf16* gA = A + (size_t)(m0 + rA) * Kk + (kt << 5) + sgA * 16;
        async_cp16(gA,     &sA[buf][rA * 40 + sgA * 16]);
        async_cp16(gA + 8, &sA[buf][rA * 40 + sgA * 16 + 8]);
        const __bf16* gB = BT + (size_t)(n0 + tid) * Kk + (kt << 5);
        #pragma unroll
        for (int c = 0; c < 4; ++c)
            async_cp16(gB + c * 8, &sB[buf][tid * 40 + c * 8]);
    };

    issue(0, 0);
    for (int kt = 0; kt < nt; ++kt) {
        const int cur = kt & 1;
        async_wait0();                       // this wave's tiles are in LDS
        __syncthreads();                     // everyone's tiles are in LDS
        if (kt + 1 < nt) issue(cur ^ 1, kt + 1);   // overlap with compute

        const __bf16* pA = sA[cur];
        const __bf16* pB = sB[cur];
#else
    __shared__ __bf16 sA1[64 * 40];
    __shared__ __bf16 sB1[128 * 40];

    for (int kt = 0; kt < nt; ++kt) {
        const __bf16* gA = A + (size_t)(m0 + rA) * Kk + (kt << 5) + sgA * 16;
        const __bf16* gB = BT + (size_t)(n0 + tid) * Kk + (kt << 5);
        if (kt + 1 < nt) {                   // gfx1250 global_prefetch_b8
            __builtin_prefetch(gA + 32, 0, 1);
            __builtin_prefetch(gB + 32, 0, 1);
        }
        uint4 a0 = *(const uint4*)gA;
        uint4 a1 = *(const uint4*)(gA + 8);
        uint4 b0 = *(const uint4*)gB;
        uint4 b1 = *(const uint4*)(gB + 8);
        uint4 b2 = *(const uint4*)(gB + 16);
        uint4 b3 = *(const uint4*)(gB + 24);
        __syncthreads();
        *(uint4*)&sA1[rA*40 + sgA*16]     = a0;
        *(uint4*)&sA1[rA*40 + sgA*16 + 8] = a1;
        *(uint4*)&sB1[tid*40]      = b0;
        *(uint4*)&sB1[tid*40 +  8] = b1;
        *(uint4*)&sB1[tid*40 + 16] = b2;
        *(uint4*)&sB1[tid*40 + 24] = b3;
        __syncthreads();

        const __bf16* pA = sA1;
        const __bf16* pB = sB1;
#endif
        const int kh = lane >> 4;
        FragBF fa[2], fb[4];
        // A 16x32 frag: lanes 0-15 rows M=0..15 K{0-7,16-23}; lanes 16-31 K{8-15,24-31}
        #pragma unroll
        for (int i = 0; i < 2; ++i) {
            const __bf16* p = &pA[(wm + i*16 + (lane & 15)) * 40 + kh * 8];
            fa[i].q[0] = *(const uint4*)p;
            fa[i].q[1] = *(const uint4*)(p + 16);
        }
        // B 32x16 frag: lane = column; lanes 0-15 K{0-7,16-23}; lanes 16-31 K{8-15,24-31}
        #pragma unroll
        for (int j = 0; j < 4; ++j) {
            const __bf16* p = &pB[(wn + j*16 + (lane & 15)) * 40 + kh * 8];
            fb[j].q[0] = *(const uint4*)p;
            fb[j].q[1] = *(const uint4*)(p + 16);
        }
        #pragma unroll
        for (int i = 0; i < 2; ++i)
            #pragma unroll
            for (int j = 0; j < 4; ++j)
                acc[i][j] = __builtin_amdgcn_wmma_f32_16x16x32_bf16(
                    false, fa[i].v, false, fb[j].v, (short)0, acc[i][j],
                    false, false);
        __syncthreads();
    }

    int cnt = 0;
    if constexpr (EPI == 2) {
        cnt = *cnt_ptr;
        if (cnt > CAPTY) cnt = CAPTY;
    }

    #pragma unroll
    for (int i = 0; i < 2; ++i) {
        #pragma unroll
        for (int j = 0; j < 4; ++j) {
            #pragma unroll
            for (int v = 0; v < 8; ++v) {
                // C/D layout: VGPR v -> row v (lanes 0-15) / v+8 (lanes 16-31)
                int gr = m0 + wm + i*16 + ((lane >> 4) << 3) + v;
                int gc = n0 + wn + j*16 + (lane & 15);
                float val = acc[i][j][v] + bias[gc];
                if constexpr (EPI == 0) {
                    Cf[(size_t)gr * Nn + gc] = val;
                } else if constexpr (EPI == 1) {
                    Cb[(size_t)gr * Nn + gc] = (__bf16)gelu_exact(val);
                } else {
                    if (gr < cnt) {
                        int tok = slot_tok[gr];
                        Cf[(size_t)tok * Nn + gc] = val;   // unique per token
                    }
                }
            }
        }
    }
}

// ============================================================================
// LayerNorm (+ optional exact GELU) over rows of DFF, one wave32 per row.
// ============================================================================
template<bool GELU>
__global__ __launch_bounds__(128)
void ln_act(const float* __restrict__ X, const float* __restrict__ g,
            const float* __restrict__ b, __bf16* __restrict__ Y, int rows)
{
    int row = blockIdx.x * 4 + (threadIdx.x >> 5);
    if (row >= rows) return;
    int lane = threadIdx.x & 31;
    const float* x = X + (size_t)row * DFF + lane * 16;
    float v[16], s = 0.f, s2 = 0.f;
    #pragma unroll
    for (int j = 0; j < 16; ++j) { v[j] = x[j]; s += v[j]; s2 += v[j]*v[j]; }
    #pragma unroll
    for (int off = 16; off > 0; off >>= 1) {
        s  += __shfl_xor(s,  off, 32);
        s2 += __shfl_xor(s2, off, 32);
    }
    float m    = s  * (1.0f / DFF);
    float var  = s2 * (1.0f / DFF) - m * m;
    float rstd = rsqrtf(var + LN_EPS);
    __bf16* y = Y + (size_t)row * DFF + lane * 16;
    #pragma unroll
    for (int j = 0; j < 16; ++j) {
        int c = lane * 16 + j;
        float t = (v[j] - m) * rstd * g[c] + b[c];
        if (GELU) t = gelu_exact(t);
        y[j] = (__bf16)t;
    }
}

// ============================================================================
// Router: logits = g1 @ Wg2 + bg2, top-1, capacity slot assignment.
// ============================================================================
__global__ __launch_bounds__(256)
void route_kernel(const __bf16* __restrict__ g1, const float* __restrict__ Wg2,
                  const float* __restrict__ bg2, int* __restrict__ counts,
                  int* __restrict__ slot_tok)
{
    __shared__ float sW[GATEH][KEXP];
    __shared__ float sb[KEXP];
    int tid = threadIdx.x;
    #pragma unroll
    for (int j = 0; j < KEXP; ++j) sW[tid][j] = Wg2[tid * KEXP + j];
    if (tid < KEXP) sb[tid] = bg2[tid];
    __syncthreads();

    int tok = blockIdx.x * 256 + tid;
    float acc[KEXP];
    #pragma unroll
    for (int k = 0; k < KEXP; ++k) acc[k] = sb[k];
    const __bf16* gr = g1 + (size_t)tok * GATEH;
    for (int j = 0; j < GATEH; ++j) {
        float gv = (float)gr[j];
        #pragma unroll
        for (int k = 0; k < KEXP; ++k) acc[k] += gv * sW[j][k];
    }
    int best = 0; float bv = acc[0];
    #pragma unroll
    for (int k = 1; k < KEXP; ++k)
        if (acc[k] > bv) { bv = acc[k]; best = k; }   // first-max like argmax
    int pos = atomicAdd(&counts[best], 1);
    if (pos < CAPTY) slot_tok[best * CAPTY + pos] = tok;
}

// ============================================================================
// Gather tokens for one expert: xb[slot] = bf16(expert_LN(base[token])), else 0
// ============================================================================
__global__ __launch_bounds__(128)
void expert_gather(const float* __restrict__ base, const float* __restrict__ eg,
                   const float* __restrict__ eb, const int* __restrict__ slot_tok,
                   const int* __restrict__ cnt_ptr, __bf16* __restrict__ xb)
{
    int slot = blockIdx.x * 4 + (threadIdx.x >> 5);
    int lane = threadIdx.x & 31;
    int cnt = *cnt_ptr; if (cnt > CAPTY) cnt = CAPTY;
    __bf16* y = xb + (size_t)slot * DFF + lane * 16;
    if (slot >= cnt) {
        #pragma unroll
        for (int j = 0; j < 16; ++j) y[j] = (__bf16)0.0f;
        return;
    }
    int tok = slot_tok[slot];
    const float* x = base + (size_t)tok * DFF + lane * 16;
    float v[16], s = 0.f, s2 = 0.f;
    #pragma unroll
    for (int j = 0; j < 16; ++j) { v[j] = x[j]; s += v[j]; s2 += v[j]*v[j]; }
    #pragma unroll
    for (int off = 16; off > 0; off >>= 1) {
        s  += __shfl_xor(s,  off, 32);
        s2 += __shfl_xor(s2, off, 32);
    }
    float m    = s  * (1.0f / DFF);
    float var  = s2 * (1.0f / DFF) - m * m;
    float rstd = rsqrtf(var + LN_EPS);
    #pragma unroll
    for (int j = 0; j < 16; ++j) {
        int c = lane * 16 + j;
        y[j] = (__bf16)((v[j] - m) * rstd * eg[c] + eb[c]);
    }
}

// ---- small elementwise helpers ----
__global__ void convw_kernel(const float* __restrict__ in, __bf16* __restrict__ out,
                             int Kk, long total)   // out[n*Kk+k] = bf16(in[k*Nn+n])
{
    long i = (long)blockIdx.x * blockDim.x + threadIdx.x;
    if (i >= total) return;
    long Nn = total / Kk;
    long n = i / Kk, k = i % Kk;
    out[i] = (__bf16)in[k * Nn + n];
}

__global__ void conv_concat_kernel(const float* __restrict__ h2d,
                                   const float* __restrict__ h3d,
                                   __bf16* __restrict__ xin, long total)
{
    long i = (long)blockIdx.x * blockDim.x + threadIdx.x;
    if (i >= total) return;
    long t = i >> 9; int c = (int)(i & 511);
    float v = (c < 256) ? h2d[t * 256 + c] : h3d[t * 256 + (c - 256)];
    xin[i] = (__bf16)v;
}

__global__ void sum_convert_kernel(const float* __restrict__ a,
                                   const float* __restrict__ b,
                                   __bf16* __restrict__ y, long total)
{
    long i = (long)blockIdx.x * blockDim.x + threadIdx.x;
    if (i >= total) return;
    y[i] = (__bf16)(a[i] + b[i]);
}

__global__ void zero_u32_kernel(uint32_t* __restrict__ p, long n)
{
    long stride = (long)gridDim.x * blockDim.x;
    for (long i = (long)blockIdx.x * blockDim.x + threadIdx.x; i < n; i += stride)
        p[i] = 0u;
}

// ============================================================================
extern "C" void kernel_launch(void* const* d_in, const int* in_sizes, int n_in,
                              void* d_out, int out_size, void* d_ws, size_t ws_size,
                              hipStream_t stream)
{
    (void)in_sizes; (void)n_in; (void)out_size; (void)ws_size;
    const float* h2d    = (const float*)d_in[0];
    const float* h3d    = (const float*)d_in[1];
    const float* Wi     = (const float*)d_in[2];
    const float* bi     = (const float*)d_in[3];
    const float* sh_g   = (const float*)d_in[4];
    const float* sh_b   = (const float*)d_in[5];
    const float* Ws     = (const float*)d_in[6];
    const float* bs     = (const float*)d_in[7];
    const float* g_ln_g = (const float*)d_in[8];
    const float* g_ln_b = (const float*)d_in[9];
    const float* Wg1    = (const float*)d_in[10];
    const float* bg1    = (const float*)d_in[11];
    const float* Wg2    = (const float*)d_in[12];
    const float* bg2    = (const float*)d_in[13];
    const float* e_ln_g = (const float*)d_in[14];
    const float* e_ln_b = (const float*)d_in[15];
    const float* We1    = (const float*)d_in[16];
    const float* be1    = (const float*)d_in[17];
    const float* We2    = (const float*)d_in[18];
    const float* be2    = (const float*)d_in[19];
    const float* Wo     = (const float*)d_in[20];
    const float* bo     = (const float*)d_in[21];
    float* out = (float*)d_out;

    char* ws = (char*)d_ws;
    size_t off = 0;
    auto alloc = [&](size_t bytes) -> void* {
        void* p = ws + off;
        off += (bytes + 255) & ~(size_t)255;
        return p;
    };

    // bf16 transposed weights
    __bf16* wWiT  = (__bf16*)alloc((size_t)DFF * DFF  * 2);
    __bf16* wWsT  = (__bf16*)alloc((size_t)DFF * DFF  * 2);
    __bf16* wWg1T = (__bf16*)alloc((size_t)DFF * GATEH * 2);
    __bf16* wWe1T = (__bf16*)alloc((size_t)KEXP * DFF * HIDD * 2);
    __bf16* wWe2T = (__bf16*)alloc((size_t)KEXP * HIDD * DFF * 2);
    __bf16* wWoT  = (__bf16*)alloc((size_t)DFF * DFF  * 2);
    // activations
    __bf16* xin   = (__bf16*)alloc((size_t)T_TOK * DFF * 2);   // concat in; reused as final sum
    float*  bufA  = (float*) alloc((size_t)T_TOK * DFF * 4);   // x, then out_f
    float*  bufB  = (float*) alloc((size_t)T_TOK * DFF * 4);   // base
    __bf16* bufC  = (__bf16*)alloc((size_t)T_TOK * DFF * 2);   // xa, then ln2
    __bf16* bufD  = (__bf16*)alloc((size_t)T_TOK * GATEH * 2); // g1
    __bf16* xb    = (__bf16*)alloc((size_t)CAPTY * DFF * 2);   // per-expert input
    __bf16* h1    = (__bf16*)alloc((size_t)CAPTY * HIDD * 2);  // per-expert hidden
    int*    slotT = (int*)   alloc((size_t)KEXP * CAPTY * 4);
    int*    cnts  = (int*)   alloc((size_t)KEXP * 4);

    const long TD  = (long)T_TOK * DFF;
    dim3 blk256(256);

    // --- weight conversion (f32 [K,N] -> bf16 [N,K]) ---
    {
        long n1 = (long)DFF * DFF;
        convw_kernel<<<(n1 + 255) / 256, blk256, 0, stream>>>(Wi,  wWiT,  DFF, n1);
        convw_kernel<<<(n1 + 255) / 256, blk256, 0, stream>>>(Ws,  wWsT,  DFF, n1);
        convw_kernel<<<(n1 + 255) / 256, blk256, 0, stream>>>(Wo,  wWoT,  DFF, n1);
        long n2 = (long)DFF * GATEH;
        convw_kernel<<<(n2 + 255) / 256, blk256, 0, stream>>>(Wg1, wWg1T, DFF, n2);
        long ne = (long)DFF * HIDD;
        for (int k = 0; k < KEXP; ++k) {
            convw_kernel<<<(ne + 255) / 256, blk256, 0, stream>>>(
                We1 + (size_t)k * ne, wWe1T + (size_t)k * ne, DFF, ne);
            convw_kernel<<<(ne + 255) / 256, blk256, 0, stream>>>(
                We2 + (size_t)k * ne, wWe2T + (size_t)k * ne, HIDD, ne);
        }
    }

    // --- concat + bf16 convert input; zero routing counters ---
    conv_concat_kernel<<<(TD + 255) / 256, blk256, 0, stream>>>(h2d, h3d, xin, TD);
    zero_u32_kernel<<<1, 64, 0, stream>>>((uint32_t*)cnts, KEXP);

    // --- x = concat @ Wi + bi ---
    gemm_bf16<0><<<dim3(T_TOK / 64, DFF / 128), 128, 0, stream>>>(
        xin, wWiT, bi, bufA, nullptr, nullptr, nullptr, T_TOK, DFF, DFF);
    // --- xa = gelu(ln(x)) ---
    ln_act<true><<<T_TOK / 4, 128, 0, stream>>>(bufA, sh_g, sh_b, bufC, T_TOK);
    // bufA now free -> becomes out_f (zero it)
    zero_u32_kernel<<<2048, 256, 0, stream>>>((uint32_t*)bufA, TD);
    // --- base = xa @ Ws + bs ---
    gemm_bf16<0><<<dim3(T_TOK / 64, DFF / 128), 128, 0, stream>>>(
        bufC, wWsT, bs, bufB, nullptr, nullptr, nullptr, T_TOK, DFF, DFF);
    // --- ln2 = ln(base) ---
    ln_act<false><<<T_TOK / 4, 128, 0, stream>>>(bufB, g_ln_g, g_ln_b, bufC, T_TOK);
    // --- g1 = gelu(ln2 @ Wg1 + bg1) ---
    gemm_bf16<1><<<dim3(T_TOK / 64, GATEH / 128), 128, 0, stream>>>(
        bufC, wWg1T, bg1, nullptr, bufD, nullptr, nullptr, T_TOK, GATEH, DFF);
    // --- route: logits, top-1, capacity slots ---
    route_kernel<<<T_TOK / 256, 256, 0, stream>>>(bufD, Wg2, bg2, cnts, slotT);

    // --- experts, one at a time ---
    for (int k = 0; k < KEXP; ++k) {
        expert_gather<<<CAPTY / 4, 128, 0, stream>>>(
            bufB, e_ln_g + (size_t)k * DFF, e_ln_b + (size_t)k * DFF,
            slotT + (size_t)k * CAPTY, cnts + k, xb);
        gemm_bf16<1><<<dim3(CAPTY / 64, HIDD / 128), 128, 0, stream>>>(
            xb, wWe1T + (size_t)k * DFF * HIDD, be1 + (size_t)k * HIDD,
            nullptr, h1, nullptr, nullptr, CAPTY, HIDD, DFF);
        gemm_bf16<2><<<dim3(CAPTY / 64, DFF / 128), 128, 0, stream>>>(
            h1, wWe2T + (size_t)k * HIDD * DFF, be2 + (size_t)k * DFF,
            bufA, nullptr, slotT + (size_t)k * CAPTY, cnts + k, CAPTY, DFF, HIDD);
    }

    // --- out = (out_f + base) @ Wo + bo ---
    sum_convert_kernel<<<(TD + 255) / 256, blk256, 0, stream>>>(bufA, bufB, xin, TD);
    gemm_bf16<0><<<dim3(T_TOK / 64, DFF / 128), 128, 0, stream>>>(
        xin, wWoT, bo, out, nullptr, nullptr, nullptr, T_TOK, DFF, DFF);
}
